// ViT5MSDeformAttnTransformerEncoderOnly_23948737642742
// MI455X (gfx1250) — matrix operand order repaired and checked by reading, not compiled, not run on hardware
//
#include <hip/hip_runtime.h>
#include <hip/hip_bf16.h>

// ---------------- types ----------------
typedef __attribute__((ext_vector_type(16))) _Float16 v16h;
typedef __attribute__((ext_vector_type(8)))  _Float16 v8h;
typedef __attribute__((ext_vector_type(8)))  float    v8f;

#define NTOK 21760
#define DMODEL 256
#define BSZ 2
#define NHEAD 8
#define NLVL 4
#define NPNT 4
#define DFFN 1024
#define NLAYERS 6

// ---------------- WMMA fragment loads (CDNA5 wave32 layouts) ----------------
// A-matrix 16x32 f16: lane holds row m = lane&15.
// lanes 0-15: K halves {k0+0..7, k0+16..23}; lanes 16-31: {k0+8..15, k0+24..31}
__device__ __forceinline__ v16h load_a_frag(const _Float16* __restrict__ A, int lda,
                                            int m0, int k0, int lane) {
  int m  = m0 + (lane & 15);
  int kb = k0 + ((lane & 16) >> 1);   // +8 for hi half-wave
  const _Float16* p = A + (size_t)m * lda + kb;
  v8h lo = *(const v8h*)(p);
  v8h hi = *(const v8h*)(p + 16);
  return __builtin_shufflevector(lo, hi, 0,1,2,3,4,5,6,7,8,9,10,11,12,13,14,15);
}

// B-matrix 32x16 f16: lane holds column n = lane&15.
// lanes 0-15: K {k0+0..7, k0+16..23}; lanes 16-31: {k0+8..15, k0+24..31}
// B[k][n] = W[n][k] with W row-major [N,K] -> two 16B loads.
__device__ __forceinline__ v16h load_b_frag(const _Float16* __restrict__ W, int ldw,
                                            int n0, int k0, int lane) {
  int n  = n0 + (lane & 15);
  int kb = k0 + ((lane & 16) >> 1);   // +8 for hi half-wave
  const _Float16* p = W + (size_t)n * ldw + kb;
  v8h lo = *(const v8h*)(p);
  v8h hi = *(const v8h*)(p + 16);
  return __builtin_shufflevector(lo, hi, 0,1,2,3,4,5,6,7,8,9,10,11,12,13,14,15);
}

// ---------------- GEMM: C[M,N] = A[M,K] * W[N,K]^T + bias ----------------
// Block: 4 waves, 128(M) x 64(N) tile; each wave owns a 32(M) x 64(N) strip:
// per k-step 2 A frags + 4 B frags feed 8 WMMAs (B traffic amortized over 2 M-tiles).
// EPI==0: Cf = result
// EPI==3: SwiGLU gate epilogue: Ch = (f16)( silu(Cf_in) * result )   (Cf read-only)
template<int EPI>
__global__ __launch_bounds__(128)
void gemm_wmma(const _Float16* __restrict__ A, const _Float16* __restrict__ W,
               const float* __restrict__ bias, float* __restrict__ Cf,
               _Float16* __restrict__ Ch, int M, int Nn, int K) {
  const int lane = threadIdx.x & 31;
  const int wave = threadIdx.x >> 5;
  const int m0 = blockIdx.x * 128 + wave * 32;   // wave strip: rows [m0, m0+32)
  const int nb = blockIdx.y * 64;

  v8f acc[2][4];
#pragma unroll
  for (int i = 0; i < 2; ++i)
#pragma unroll
    for (int j = 0; j < 4; ++j) acc[i][j] = {};

  for (int k0 = 0; k0 < K; k0 += 32) {
    v16h a0 = load_a_frag(A, K, m0,      k0, lane);
    v16h a1 = load_a_frag(A, K, m0 + 16, k0, lane);
#pragma unroll
    for (int j = 0; j < 4; ++j) {
      v16h b = load_b_frag(W, K, nb + j * 16, k0, lane);
      acc[0][j] = __builtin_amdgcn_wmma_f32_16x16x32_f16(
          false, a0, false, b, (short)0, acc[0][j], false, false);
      acc[1][j] = __builtin_amdgcn_wmma_f32_16x16x32_f16(
          false, a1, false, b, (short)0, acc[1][j], false, false);
    }
  }

  // C/D layout: acc[i][j] element r -> (m = m0 + 16*i + r + 8*hi,
  //                                     n = nb + 16*j + (lane&15))
  const int nlo   = nb + (lane & 15);
  const int mbase = m0 + ((lane & 16) >> 1);
#pragma unroll
  for (int i = 0; i < 2; ++i) {
#pragma unroll
    for (int j = 0; j < 4; ++j) {
      int n = nlo + j * 16;
      float bv = bias ? bias[n] : 0.0f;
#pragma unroll
      for (int r = 0; r < 8; ++r) {
        int m = mbase + i * 16 + r;
        float v = acc[i][j][r] + bv;
        size_t idx = (size_t)m * Nn + n;
        if (EPI == 0) {
          Cf[idx] = v;
        } else {                       // SwiGLU: silu(h1) * h3
          float h1v = Cf[idx];
          float sg  = h1v / (1.0f + __expf(-h1v));
          Ch[idx]   = (_Float16)(sg * v);
        }
      }
    }
  }
}

// ---------------- pack: NCHW level -> [BS, N, D] tokens (+ pos + level_embed) --------
__global__ void pack_kernel(const float* __restrict__ src, const float* __restrict__ pos,
                            const float* __restrict__ lemb,
                            float* __restrict__ qd, float* __restrict__ pd,
                            int HW, int base) {
  int i = blockIdx.x * blockDim.x + threadIdx.x;
  int total = BSZ * HW * DMODEL;
  if (i >= total) return;
  int d = i & 255;
  int s = (i >> 8) % HW;
  int b = i / (HW * DMODEL);
  size_t srci = ((size_t)b * DMODEL + d) * HW + s;
  size_t dsti = ((size_t)b * NTOK + base + s) * (size_t)DMODEL + d;
  qd[dsti] = src[srci];
  pd[dsti] = pos[srci] + lemb[d];
}

// ---------------- f32 -> f16 ----------------
__global__ void cvt_f16_kernel(const float* __restrict__ s, _Float16* __restrict__ d, size_t n) {
  size_t i = (size_t)blockIdx.x * blockDim.x + threadIdx.x;
  if (i < n) d[i] = (_Float16)s[i];
}

// ---------------- RMSNorm (one wave32 per row of D=256) ----------------
__global__ void rmsnorm_kernel(const float* __restrict__ x, const float* __restrict__ g,
                               const float* __restrict__ posf,
                               _Float16* __restrict__ n16, _Float16* __restrict__ qin16,
                               int rows) {
  int wave = threadIdx.x >> 5;
  int lane = threadIdx.x & 31;
  int row  = blockIdx.x * (blockDim.x >> 5) + wave;
  if (row >= rows) return;
  const float* xr = x + (size_t)row * DMODEL + lane * 8;
  float v[8];
  float ss = 0.f;
#pragma unroll
  for (int i = 0; i < 8; ++i) { v[i] = xr[i]; ss += v[i] * v[i]; }
#pragma unroll
  for (int m = 16; m >= 1; m >>= 1) ss += __shfl_xor(ss, m, 32);
  float inv = rsqrtf(ss * (1.0f / DMODEL) + 1e-6f);
  int d0 = lane * 8;
  size_t roff = (size_t)row * DMODEL + d0;
#pragma unroll
  for (int i = 0; i < 8; ++i) {
    float nv = v[i] * inv * g[d0 + i];
    n16[roff + i] = (_Float16)nv;
    if (qin16) qin16[roff + i] = (_Float16)(nv + posf[roff + i]);
  }
}

// ---------------- softmax over 16 (L*P) per (b,q,h), in place ----------------
__global__ void softmax16_kernel(float* __restrict__ aw, int units) {
  int u = blockIdx.x * blockDim.x + threadIdx.x;
  if (u >= units) return;
  float* p = aw + (size_t)(u >> 3) * 128 + (size_t)(u & 7) * 16;
  float mx = -1e30f;
#pragma unroll
  for (int i = 0; i < 16; ++i) mx = fmaxf(mx, p[i]);
  float e[16], sum = 0.f;
#pragma unroll
  for (int i = 0; i < 16; ++i) { e[i] = __expf(p[i] - mx); sum += e[i]; }
  float inv = 1.0f / sum;
#pragma unroll
  for (int i = 0; i < 16; ++i) p[i] = e[i] * inv;
}

// ---------------- deformable bilinear sampling ----------------
// unit = (b,q,h), lane = channel dh (0..31). 8 units per 256-thread block.
__global__ void msda_sample_kernel(const float* __restrict__ value,
                                   const float* __restrict__ off,
                                   const float* __restrict__ aw,
                                   float* __restrict__ samp,
                                   _Float16* __restrict__ samp16) {
  const int Hs[4] = {128, 64, 32, 16};
  const int Ws[4] = {128, 64, 32, 16};
  const int St[4] = {0, 16384, 20480, 21504};
  int unit = blockIdx.x * 8 + (threadIdx.x >> 5);
  int lane = threadIdx.x & 31;
  if (unit >= BSZ * NTOK * NHEAD) return;
  int h = unit & 7;
  int q = (unit >> 3) % NTOK;
  int b = unit / (NTOK * NHEAD);

  int lvl, s;
  if (q < 16384)      { lvl = 0; s = q; }
  else if (q < 20480) { lvl = 1; s = q - 16384; }
  else if (q < 21504) { lvl = 2; s = q - 20480; }
  else                { lvl = 3; s = q - 21504; }
  float rx = ((s % Ws[lvl]) + 0.5f) / (float)Ws[lvl];
  float ry = ((s / Ws[lvl]) + 0.5f) / (float)Hs[lvl];

  size_t rowi = (size_t)b * NTOK + q;
  const float* offp = off + rowi * 256 + h * 32;   // [NH, L, P, 2]
  const float* awp  = aw  + rowi * 128 + h * 16;   // [NH, L, P]

  float acc = 0.f;
#pragma unroll
  for (int l = 0; l < 4; ++l) {
    int Hl = Hs[l], Wl = Ws[l];
    const float* vbase = value + (((size_t)b * NTOK + St[l]) * DMODEL) + h * 32 + lane;
#pragma unroll
    for (int p = 0; p < 4; ++p) {
      float ox = offp[l * 8 + p * 2 + 0];
      float oy = offp[l * 8 + p * 2 + 1];
      float x = (rx + ox / (float)Wl) * (float)Wl - 0.5f;
      float y = (ry + oy / (float)Hl) * (float)Hl - 0.5f;
      float x0 = floorf(x), y0 = floorf(y);
      float w = awp[l * 4 + p];
      float sv = 0.f;
#pragma unroll
      for (int c = 0; c < 4; ++c) {
        float xi = x0 + (float)(c & 1);
        float yi = y0 + (float)(c >> 1);
        float wt = (1.f - fabsf(x - xi)) * (1.f - fabsf(y - yi));
        if (xi >= 0.f && xi < (float)Wl && yi >= 0.f && yi < (float)Hl) {
          int idx = (int)yi * Wl + (int)xi;
          sv += wt * vbase[(size_t)idx * DMODEL];
        }
      }
      acc += w * sv;
    }
  }
  size_t oidx = rowi * DMODEL + h * 32 + lane;
  samp[oidx]   = acc;
  samp16[oidx] = (_Float16)acc;
}

// ---------------- residual: q += ls[d] * a ----------------
__global__ void axpy_ls_kernel(float* __restrict__ q, const float* __restrict__ a,
                               const float* __restrict__ ls, size_t n) {
  size_t i = (size_t)blockIdx.x * blockDim.x + threadIdx.x;
  if (i >= n) return;
  q[i] += ls[i & 255] * a[i];
}

// ---------------- final output (+ int tail: spatial_shapes, level_start_index) -------
__global__ void finalize_kernel(const float* __restrict__ q, float* __restrict__ out, size_t md) {
  size_t i = (size_t)blockIdx.x * blockDim.x + threadIdx.x;
  if (i < md) out[i] = q[i];
  if (i == 0) {
    int* t = (int*)(out + md);
    t[0] = 128; t[1] = 128; t[2] = 64; t[3] = 64;
    t[4] = 32;  t[5] = 32;  t[6] = 16; t[7] = 16;
    t[8] = 0; t[9] = 16384; t[10] = 20480; t[11] = 21504;
  }
}

// ================= host side =================
static inline int cdiv(size_t a, size_t b) { return (int)((a + b - 1) / b); }

extern "C" void kernel_launch(void* const* d_in, const int* in_sizes, int n_in,
                              void* d_out, int out_size, void* d_ws, size_t ws_size,
                              hipStream_t stream) {
  (void)in_sizes; (void)n_in; (void)out_size; (void)ws_size;

  const float* src[4] = {(const float*)d_in[0], (const float*)d_in[2],
                         (const float*)d_in[4], (const float*)d_in[6]};
  const float* pos[4] = {(const float*)d_in[1], (const float*)d_in[3],
                         (const float*)d_in[5], (const float*)d_in[7]};
  const float* level_embed = (const float*)d_in[8];
  const float* W_off = (const float*)d_in[9];   const float* b_off = (const float*)d_in[10];
  const float* W_attn= (const float*)d_in[11];  const float* b_attn= (const float*)d_in[12];
  const float* W_val = (const float*)d_in[13];  const float* b_val = (const float*)d_in[14];
  const float* W_out = (const float*)d_in[15];  const float* b_out = (const float*)d_in[16];
  const float* g_na  = (const float*)d_in[17];  const float* g_nf  = (const float*)d_in[18];
  const float* ls_a  = (const float*)d_in[19];  const float* ls_f  = (const float*)d_in[20];
  const float* W1 = (const float*)d_in[21];     const float* b1 = (const float*)d_in[22];
  const float* W3 = (const float*)d_in[23];     const float* b3 = (const float*)d_in[24];
  const float* W2 = (const float*)d_in[25];     const float* b2 = (const float*)d_in[26];

  const size_t MD = (size_t)BSZ * NTOK * DMODEL;   // 11,141,120
  const int    Mrows = BSZ * NTOK;                 // 43,520 (divisible by 128)

  // ---- workspace layout ----
  char* wp = (char*)d_ws;
  auto alloc = [&](size_t bytes) -> char* {
    char* r = wp; wp += (bytes + 255) & ~(size_t)255; return r;
  };
  float*    q_buf   = (float*)   alloc(MD * 4);
  float*    posf    = (float*)   alloc(MD * 4);
  _Float16* n16     = (_Float16*)alloc(MD * 2);
  _Float16* qin16   = (_Float16*)alloc(MD * 2);
  float*    value   = (float*)   alloc(MD * 4);
  float*    offb    = (float*)   alloc(MD * 4);
  float*    awb     = (float*)   alloc(MD / 2 * 4);
  float*    samp    = (float*)   alloc(MD * 4);
  _Float16* samp16  = (_Float16*)alloc(MD * 2);
  float*    tmp     = (float*)   alloc(MD * 4);
  float*    h1      = (float*)   alloc(MD * 16);   // [Mrows, 1024] f32
  _Float16* g16     = (_Float16*)alloc(MD * 8);    // [Mrows, 1024] f16
  _Float16* wo16    = (_Float16*)alloc((size_t)NLAYERS * 256 * 256 * 2);
  _Float16* wa16    = (_Float16*)alloc((size_t)NLAYERS * 128 * 256 * 2);
  _Float16* wv16    = (_Float16*)alloc((size_t)NLAYERS * 256 * 256 * 2);
  _Float16* wp16    = (_Float16*)alloc((size_t)NLAYERS * 256 * 256 * 2);
  _Float16* w1_16   = (_Float16*)alloc((size_t)NLAYERS * 1024 * 256 * 2);
  _Float16* w3_16   = (_Float16*)alloc((size_t)NLAYERS * 1024 * 256 * 2);
  _Float16* w2_16   = (_Float16*)alloc((size_t)NLAYERS * 256 * 1024 * 2);

  // ---- weight conversion f32 -> f16 ----
  auto cvt = [&](const float* s, _Float16* d, size_t n) {
    cvt_f16_kernel<<<cdiv(n, 256), 256, 0, stream>>>(s, d, n);
  };
  cvt(W_off,  wo16,  (size_t)NLAYERS * 256 * 256);
  cvt(W_attn, wa16,  (size_t)NLAYERS * 128 * 256);
  cvt(W_val,  wv16,  (size_t)NLAYERS * 256 * 256);
  cvt(W_out,  wp16,  (size_t)NLAYERS * 256 * 256);
  cvt(W1,     w1_16, (size_t)NLAYERS * 1024 * 256);
  cvt(W3,     w3_16, (size_t)NLAYERS * 1024 * 256);
  cvt(W2,     w2_16, (size_t)NLAYERS * 256 * 1024);

  // ---- pack levels into token buffers ----
  {
    const int HWs[4] = {128 * 128, 64 * 64, 32 * 32, 16 * 16};
    int base = 0;
    for (int l = 0; l < 4; ++l) {
      int total = BSZ * HWs[l] * DMODEL;
      pack_kernel<<<cdiv(total, 256), 256, 0, stream>>>(
          src[l], pos[l], level_embed + (size_t)l * DMODEL, q_buf, posf, HWs[l], base);
      base += HWs[l];
    }
  }

  // ---- encoder layers ----
  for (int i = 0; i < NLAYERS; ++i) {
    const _Float16* Wo = wo16  + (size_t)i * 256 * 256;
    const _Float16* Wa = wa16  + (size_t)i * 128 * 256;
    const _Float16* Wv = wv16  + (size_t)i * 256 * 256;
    const _Float16* Wp = wp16  + (size_t)i * 256 * 256;
    const _Float16* Wf1= w1_16 + (size_t)i * 1024 * 256;
    const _Float16* Wf3= w3_16 + (size_t)i * 1024 * 256;
    const _Float16* Wf2= w2_16 + (size_t)i * 256 * 1024;

    // attention pre-norm: n16 = rmsnorm(q)*g ; qin16 = n + posf
    rmsnorm_kernel<<<cdiv(Mrows, 8), 256, 0, stream>>>(
        q_buf, g_na + (size_t)i * 256, posf, n16, qin16, Mrows);

    // value / offsets / attention-logit GEMMs (WMMA)
    gemm_wmma<0><<<dim3(Mrows / 128, 256 / 64), 128, 0, stream>>>(
        n16, Wv, b_val + (size_t)i * 256, value, (_Float16*)nullptr, Mrows, 256, 256);
    gemm_wmma<0><<<dim3(Mrows / 128, 256 / 64), 128, 0, stream>>>(
        qin16, Wo, b_off + (size_t)i * 256, offb, (_Float16*)nullptr, Mrows, 256, 256);
    gemm_wmma<0><<<dim3(Mrows / 128, 128 / 64), 128, 0, stream>>>(
        qin16, Wa, b_attn + (size_t)i * 128, awb, (_Float16*)nullptr, Mrows, 128, 256);

    softmax16_kernel<<<cdiv((size_t)Mrows * NHEAD, 256), 256, 0, stream>>>(
        awb, Mrows * NHEAD);

    msda_sample_kernel<<<cdiv((size_t)Mrows * NHEAD, 8), 256, 0, stream>>>(
        value, offb, awb, samp, samp16);

    // output projection + residual
    gemm_wmma<0><<<dim3(Mrows / 128, 256 / 64), 128, 0, stream>>>(
        samp16, Wp, b_out + (size_t)i * 256, tmp, (_Float16*)nullptr, Mrows, 256, 256);
    axpy_ls_kernel<<<cdiv(MD, 256), 256, 0, stream>>>(
        q_buf, tmp, ls_a + (size_t)i * 256, MD);

    // FFN pre-norm
    rmsnorm_kernel<<<cdiv(Mrows, 8), 256, 0, stream>>>(
        q_buf, g_nf + (size_t)i * 256, (const float*)nullptr, n16, (_Float16*)nullptr, Mrows);

    // h1 = n @ W1^T + b1 ; g16 = silu(h1) * (n @ W3^T + b3)  (fused epilogue)
    gemm_wmma<0><<<dim3(Mrows / 128, 1024 / 64), 128, 0, stream>>>(
        n16, Wf1, b1 + (size_t)i * 1024, h1, (_Float16*)nullptr, Mrows, 1024, 256);
    gemm_wmma<3><<<dim3(Mrows / 128, 1024 / 64), 128, 0, stream>>>(
        n16, Wf3, b3 + (size_t)i * 1024, h1, g16, Mrows, 1024, 256);

    // f = g @ W2^T + b2 ; q += ls_f * f
    gemm_wmma<0><<<dim3(Mrows / 128, 256 / 64), 128, 0, stream>>>(
        g16, Wf2, b2 + (size_t)i * 256, tmp, (_Float16*)nullptr, Mrows, 256, 1024);
    axpy_ls_kernel<<<cdiv(MD, 256), 256, 0, stream>>>(
        q_buf, tmp, ls_f + (size_t)i * 256, MD);
  }

  // ---- final output: q tokens + int32 tail (spatial_shapes, level_start_index) ----
  finalize_kernel<<<cdiv(MD, 256), 256, 0, stream>>>(q_buf, (float*)d_out, MD);
}
